// PVT_MDTA_68590627717640
// MI455X (gfx1250) — compile-verified, matrix operand order checked by reading, not run on hardware
//
#include <hip/hip_runtime.h>
#include <hip/hip_bf16.h>

// ---------------------------------------------------------------------------
// MDTA (transposed/channel attention) for MI455X gfx1250, wave32 + WMMA bf16.
//
// Fusion:  out = LayerNorm_W( Wo · blockdiag(softmax(QK^T/√hw)) · Wv · x + beff )
// Pass A: stream x, build Q/K tiles with v_wmma_f32_16x16x32_bf16, accumulate
//         per-head 16x16 QK^T partials (K = spatial) -> deterministic ws slots.
// Pass B: softmax + Weff_b = Wo·blockdiag(attn)·Wv (WMMA), beff_b.
// Pass C: Weff_b·x_b (WMMA) with LayerNorm fused via 16-lane shfl reductions.
// ---------------------------------------------------------------------------

typedef __attribute__((ext_vector_type(16))) __bf16 v16bf;
typedef __attribute__((ext_vector_type(8)))  __bf16 v8bf;
typedef __attribute__((ext_vector_type(8)))  float  v8f;

#define BATCH   16
#define CH      128
#define HW      16384
#define NH      8
#define SCALE   (1.0f/128.0f)     // hw^-0.5 = 16384^-0.5
#define TPOS    64                // spatial tile in pass A
#define SPLITS  32                // spatial splits per batch in pass A
#define TILES   ((HW / SPLITS) / TPOS)   // 8

// --- WMMA fragment loaders (CDNA5 ISA §7.12.2, wave32) ----------
// 16-bit A (16x32): lane l: row = l&15, lh = l>>4
//   elems 0..7  -> K = kbase + 8*lh + i
//   elems 8..15 -> K = kbase + 16 + 8*lh + (i-8)
// B (32x16) mirrored with lane -> column; we always store the B operand
// transposed ([n][k]) so the same contiguous-K loader serves both operands.
__device__ __forceinline__ v16bf frag_bf(const __bf16* block, int ld, int kbase, int lane) {
  const int r = lane & 15, lh = lane >> 4;
  const __bf16* p = block + r * ld + kbase + lh * 8;
  union { v16bf v; v8bf h[2]; } u;
  u.h[0] = *(const v8bf*)(p);        // ds_load_b128 / global_load_b128
  u.h[1] = *(const v8bf*)(p + 16);
  return u.v;
}

__device__ __forceinline__ v16bf frag_gf32(const float* block, int ld, int kbase, int lane) {
  const int r = lane & 15, lh = lane >> 4;
  const float* p = block + r * ld + kbase + lh * 8;
  union { v16bf v; __bf16 e[16]; } u;
#pragma unroll
  for (int i = 0; i < 8; ++i) { u.e[i] = (__bf16)p[i]; u.e[i + 8] = (__bf16)p[16 + i]; }
  return u.v;
}

__device__ __forceinline__ v8f wmma_bf(v16bf a, v16bf b, v8f c) {
  return __builtin_amdgcn_wmma_f32_16x16x32_bf16(false, a, false, b, (short)0, c, false, false);
}

// --- Pass A: attn_raw partials ---------------------------------------------
// grid (SPLITS, BATCH), 256 threads. wave == m-block == head.
__global__ __launch_bounds__(256) void qk_attn_kernel(
    const float* __restrict__ x,  const float* __restrict__ wq, const float* __restrict__ bq,
    const float* __restrict__ wk, const float* __restrict__ bk, float* __restrict__ attn_part)
{
  __shared__ __align__(16) __bf16 sXT[TPOS][CH + 8];   // x tile transposed [pos][cin]
  __shared__ __align__(16) __bf16 sQ[CH][TPOS + 8];
  __shared__ __align__(16) __bf16 sK[CH][TPOS + 8];

  const int tid = threadIdx.x, wave = tid >> 5, lane = tid & 31, lh = lane >> 4;
  const int b = blockIdx.y, split = blockIdx.x;
  const int pbase = split * (HW / SPLITS);

  // per-wave weight fragments for its 16 output channels (== its head)
  v16bf aq[4], ak[4];
#pragma unroll
  for (int kk = 0; kk < 4; ++kk) {
    aq[kk] = frag_gf32(wq + (size_t)wave * 16 * CH, CH, kk * 32, lane);
    ak[kk] = frag_gf32(wk + (size_t)wave * 16 * CH, CH, kk * 32, lane);
  }
  float qb[8], kb[8];
#pragma unroll
  for (int j = 0; j < 8; ++j) {
    qb[j] = bq[wave * 16 + j + 8 * lh];
    kb[j] = bk[wave * 16 + j + 8 * lh];
  }

  v8f acc = {};   // partial attn (16x16) for head == wave

  for (int t = 0; t < TILES; ++t) {
    const int p0 = pbase + t * TPOS;
    for (int idx = tid; idx < CH * TPOS; idx += 256) {      // coalesced, f32->bf16
      int c = idx >> 6, p = idx & 63;
      sXT[p][c] = (__bf16)x[((size_t)b * CH + c) * HW + p0 + p];
    }
    __syncthreads();

    // Q,K projection: M=16 (own rows), N=64, K=128
#pragma unroll
    for (int nb = 0; nb < 4; ++nb) {
      v8f cq = {}, ck = {};
#pragma unroll
      for (int kk = 0; kk < 4; ++kk) {
        v16bf bx = frag_bf(&sXT[nb * 16][0], CH + 8, kk * 32, lane);
        cq = wmma_bf(aq[kk], bx, cq);
        ck = wmma_bf(ak[kk], bx, ck);
      }
      int n = nb * 16 + (lane & 15);
#pragma unroll
      for (int j = 0; j < 8; ++j) {                // C/D layout: m = j + 8*lh
        int m = wave * 16 + j + 8 * lh;
        sQ[m][n] = (__bf16)(cq[j] + qb[j]);
        sK[m][n] = (__bf16)(ck[j] + kb[j]);
      }
    }
    // attn += Q·K^T over this tile (wave reads only its own 16-row block;
    // per-wave DS ordering makes this safe without a barrier)
    acc = wmma_bf(frag_bf(&sQ[wave * 16][0], TPOS + 8, 0,  lane),
                  frag_bf(&sK[wave * 16][0], TPOS + 8, 0,  lane), acc);
    acc = wmma_bf(frag_bf(&sQ[wave * 16][0], TPOS + 8, 32, lane),
                  frag_bf(&sK[wave * 16][0], TPOS + 8, 32, lane), acc);
    __syncthreads();
  }

  // deterministic per-split slot: [b][h][split][256]
  float* dst = attn_part + (((size_t)b * NH + wave) * SPLITS + split) * 256;
#pragma unroll
  for (int j = 0; j < 8; ++j)
    dst[(j + 8 * lh) * 16 + (lane & 15)] = acc[j];
}

// --- Pass B: softmax + Weff/beff -------------------------------------------
// grid BATCH, 256 threads.
__global__ __launch_bounds__(256) void weff_kernel(
    const float* __restrict__ attn_part, const float* __restrict__ wv, const float* __restrict__ bv,
    const float* __restrict__ wo, const float* __restrict__ bo,
    __bf16* __restrict__ weff, float* __restrict__ beff)
{
  __shared__ float sA[NH][16][16];
  __shared__ __align__(16) __bf16 sTt[CH][CH + 8];   // (blockdiag(A)·Wv)^T : [n][k]
  __shared__ float sAv[CH];

  const int tid = threadIdx.x, wave = tid >> 5, lane = tid & 31, lh = lane >> 4;
  const int b = blockIdx.x;

  if (tid < 128) {                       // reduce splits + softmax row
    int h = tid >> 4, r = tid & 15;
    const float* src = attn_part + (((size_t)b * NH + h) * SPLITS) * 256 + r * 16;
    float v[16], mx = -3.0e38f;
#pragma unroll
    for (int d = 0; d < 16; ++d) {
      float s = 0.f;
      for (int sp = 0; sp < SPLITS; ++sp) s += src[sp * 256 + d];
      v[d] = s * SCALE;  mx = fmaxf(mx, v[d]);
    }
    float s = 0.f;
#pragma unroll
    for (int d = 0; d < 16; ++d) { v[d] = __expf(v[d] - mx); s += v[d]; }
    float inv = 1.f / s;
#pragma unroll
    for (int d = 0; d < 16; ++d) sA[h][r][d] = v[d] * inv;
  }
  __syncthreads();

  // T = blockdiag(attn)·Wv (f32, stored transposed bf16);  Av = blockdiag(attn)·bv
  for (int idx = tid; idx < CH * CH; idx += 256) {
    int r = idx >> 7, j = idx & 127, h = r >> 4, c = r & 15;
    float s = 0.f;
#pragma unroll
    for (int d = 0; d < 16; ++d) s += sA[h][c][d] * wv[(h * 16 + d) * CH + j];
    sTt[j][r] = (__bf16)s;
  }
  if (tid < 128) {
    int h = tid >> 4, c = tid & 15;
    float s = 0.f;
#pragma unroll
    for (int d = 0; d < 16; ++d) s += sA[h][c][d] * bv[h * 16 + d];
    sAv[tid] = s;
  }
  __syncthreads();

  // Weff = Wo · T   (128x128x128 WMMA)
  v16bf aw[4];
#pragma unroll
  for (int kk = 0; kk < 4; ++kk)
    aw[kk] = frag_gf32(wo + (size_t)wave * 16 * CH, CH, kk * 32, lane);

  __bf16* wout = weff + (size_t)b * CH * CH;
#pragma unroll
  for (int nb = 0; nb < 8; ++nb) {
    v8f c = {};
#pragma unroll
    for (int kk = 0; kk < 4; ++kk)
      c = wmma_bf(aw[kk], frag_bf(&sTt[nb * 16][0], CH + 8, kk * 32, lane), c);
    int n = nb * 16 + (lane & 15);
#pragma unroll
    for (int j = 0; j < 8; ++j)
      wout[(wave * 16 + j + 8 * lh) * CH + n] = (__bf16)c[j];
  }

  if (tid < 128) {                       // beff = Wo·Av + bo
    float s = bo[tid];
    for (int k = 0; k < CH; ++k) s += wo[tid * CH + k] * sAv[k];
    beff[(size_t)b * CH + tid] = s;
  }
}

// --- Pass C: out = LN_W( Weff·x + beff ) -----------------------------------
// grid (HW/128, BATCH), 256 threads. Each wave owns rows [wave*16..+16) x one
// full h-line (128 w), so LayerNorm reduces over 8 acc tiles + 16 lanes.
__global__ __launch_bounds__(256) void out_ln_kernel(
    const float* __restrict__ x, const __bf16* __restrict__ weff, const float* __restrict__ beff,
    const float* __restrict__ gamma, const float* __restrict__ beta, float* __restrict__ out)
{
  __shared__ __align__(16) __bf16 sXT[128][CH + 8];   // [pos][cin]
  __shared__ float sGamma[128], sBeta[128];

  const int tid = threadIdx.x, wave = tid >> 5, lane = tid & 31, lh = lane >> 4;
  const int b = blockIdx.y;
  const int p0 = blockIdx.x * 128;

  if (tid < 128) { sGamma[tid] = gamma[tid]; sBeta[tid] = beta[tid]; }
  for (int idx = tid; idx < CH * 128; idx += 256) {
    int c = idx >> 7, p = idx & 127;
    sXT[p][c] = (__bf16)x[((size_t)b * CH + c) * HW + p0 + p];
  }

  const __bf16* wrow = weff + (size_t)b * CH * CH + (size_t)wave * 16 * CH;
  v16bf aw[4];
#pragma unroll
  for (int kk = 0; kk < 4; ++kk)
    aw[kk] = frag_bf(wrow, CH, kk * 32, lane);

  float bfv[8];
#pragma unroll
  for (int j = 0; j < 8; ++j) bfv[j] = beff[(size_t)b * CH + wave * 16 + j + 8 * lh];

  __syncthreads();

  v8f acc[8];
#pragma unroll
  for (int nb = 0; nb < 8; ++nb) {
    v8f c = {};
#pragma unroll
    for (int kk = 0; kk < 4; ++kk)
      c = wmma_bf(aw[kk], frag_bf(&sXT[nb * 16][0], CH + 8, kk * 32, lane), c);
#pragma unroll
    for (int j = 0; j < 8; ++j) c[j] += bfv[j];
    acc[nb] = c;
  }

  // fused LayerNorm over w (row m lives in one 16-lane half across 8 tiles)
#pragma unroll
  for (int j = 0; j < 8; ++j) {
    float s = 0.f, ss = 0.f;
#pragma unroll
    for (int nb = 0; nb < 8; ++nb) { float v = acc[nb][j]; s += v; ss += v * v; }
#pragma unroll
    for (int m = 1; m < 16; m <<= 1) {
      s  += __shfl_xor(s,  m, 32);
      ss += __shfl_xor(ss, m, 32);
    }
    float mu  = s  * (1.f / 128.f);
    float var = ss * (1.f / 128.f) - mu * mu;
    float rs  = rsqrtf(fmaxf(var, 0.f) + 1e-5f);
    size_t rowbase = ((size_t)b * CH + wave * 16 + j + 8 * lh) * HW + p0;
#pragma unroll
    for (int nb = 0; nb < 8; ++nb) {
      int n = nb * 16 + (lane & 15);
      out[rowbase + n] = (acc[nb][j] - mu) * rs * sGamma[n] + sBeta[n];
    }
  }
}

// ---------------------------------------------------------------------------
extern "C" void kernel_launch(void* const* d_in, const int* in_sizes, int n_in,
                              void* d_out, int out_size, void* d_ws, size_t ws_size,
                              hipStream_t stream) {
  const float* x     = (const float*)d_in[0];
  const float* wq    = (const float*)d_in[1];
  const float* bq    = (const float*)d_in[2];
  const float* wk    = (const float*)d_in[3];
  const float* bk    = (const float*)d_in[4];
  const float* wv    = (const float*)d_in[5];
  const float* bv    = (const float*)d_in[6];
  const float* wo    = (const float*)d_in[7];
  const float* bo    = (const float*)d_in[8];
  const float* gamma = (const float*)d_in[9];
  const float* beta  = (const float*)d_in[10];

  // workspace: attn partials (4 MB) | Weff bf16 (512 KB) | beff f32 (8 KB)
  float*  attn_part = (float*)d_ws;
  __bf16* weff      = (__bf16*)((char*)d_ws + (size_t)BATCH * NH * SPLITS * 256 * 4);
  float*  beff      = (float*)((char*)weff + (size_t)BATCH * CH * CH * 2);

  qk_attn_kernel<<<dim3(SPLITS, BATCH), 256, 0, stream>>>(x, wq, bq, wk, bk, attn_part);
  weff_kernel<<<BATCH, 256, 0, stream>>>(attn_part, wv, bv, wo, bo, weff, beff);
  out_ln_kernel<<<dim3(HW / 128, BATCH), 256, 0, stream>>>(x, weff, beff, gamma, beta, (float*)d_out);
}
